// EdgeUpdate_54090818126503
// MI455X (gfx1250) — matrix operand order repaired and checked by reading, not compiled, not run on hardware
//
#include <hip/hip_runtime.h>

typedef __attribute__((ext_vector_type(16))) _Float16 v16h;
typedef __attribute__((ext_vector_type(8)))  _Float16 v8h;
typedef __attribute__((ext_vector_type(4)))  _Float16 v4h;
typedef __attribute__((ext_vector_type(8)))  float    v8f;
typedef __attribute__((ext_vector_type(4)))  float    v4f;

#define RRELU_SLOPE (11.0f / 48.0f)
#define XP    104          // padded LDS row stride in halves for the 16x96 tile
#define WAVES 8            // waves per block (256 threads, wave32)

// Packed fragment layout offsets (in halves) inside d_ws:
//  L1: 12 frags (kb 0..2, nt 0..3) @ 0      (6144 halves)
//  L2:  8 frags (kb 0..1, nt 0..3) @ 6144   (4096 halves)
//  L3:  4 frags (kb 0..1, nt 0..1) @ 10240  (2048 halves)
#define L2_OFF 6144
#define L3_OFF 10240
#define W_TOTAL 12288

union V16U { v16h v; v8h h[2]; };

__device__ __forceinline__ v4h cvt4(v4f v) {
  v4h r;
  r.x = (_Float16)v.x; r.y = (_Float16)v.y;
  r.z = (_Float16)v.z; r.w = (_Float16)v.w;
  return r;
}

// rrelu(x) = max(x, slope*x) for 0 < slope < 1 (branch/cndmask-free)
__device__ __forceinline__ float rrelu(float v) {
  return fmaxf(v, v * RRELU_SLOPE);
}

// Pack W (row-major [K_total, N_total] f32) into per-lane-contiguous f16 B fragments.
// B-matrix (KxN, 16-bit, wave32) layout: lane = (N%16) + 16*g, element h<8 -> K_in_block = 8g+h,
// h>=8 -> K_in_block = 16+8g+(h-8).  Packed: frag*512 + lane*16 + h.
__global__ void pack_weights_kernel(const float* __restrict__ W1,
                                    const float* __restrict__ W2,
                                    const float* __restrict__ W3,
                                    _Float16* __restrict__ packed) {
  int idx = blockIdx.x * blockDim.x + threadIdx.x;
  if (idx >= W_TOTAL) return;
  const float* W; int Nout, kb, nt, rem;
  if (idx < L2_OFF) {                       // layer 1: K=96 (3 kb), N=64 (4 nt)
    int frag = idx / 512; rem = idx % 512;
    kb = frag / 4; nt = frag % 4; W = W1; Nout = 64;
  } else if (idx < L3_OFF) {                // layer 2: K=64 (2 kb), N=64 (4 nt)
    int i2 = idx - L2_OFF;
    int frag = i2 / 512; rem = i2 % 512;
    kb = frag / 4; nt = frag % 4; W = W2; Nout = 64;
  } else {                                  // layer 3: K=64 (2 kb), N=32 (2 nt)
    int i3 = idx - L3_OFF;
    int frag = i3 / 512; rem = i3 % 512;
    kb = frag / 2; nt = frag % 2; W = W3; Nout = 32;
  }
  int l = rem / 16, h = rem % 16;
  int K = kb * 32 + (h < 8 ? (l / 16) * 8 + h : 16 + (l / 16) * 8 + (h - 8));
  int N = nt * 16 + (l % 16);
  packed[idx] = (_Float16)W[K * Nout + N];
}

__global__ __launch_bounds__(256) void edge_mlp_kernel(
    const float* __restrict__ bonds, const int* __restrict__ ba1,
    const int* __restrict__ ba2,    const float* __restrict__ atoms,
    const _Float16* __restrict__ wpack,
    const float* __restrict__ b1, const float* __restrict__ b2,
    const float* __restrict__ b3, float* __restrict__ out, int n_tiles)
{
  __shared__ _Float16 wlds3[2048];                         // layer-3 B frags (4 KB)
  __shared__ __align__(16) char slabs[WAVES][16 * XP * 2]; // 3328 B per wave

  // Stage layer-3 fragments into LDS (4096 B = 256 uint4; one per thread).
  ((uint4*)wlds3)[threadIdx.x] = ((const uint4*)(wpack + L3_OFF))[threadIdx.x];
  __syncthreads();

  const int lane = threadIdx.x & 31;
  const int wave = threadIdx.x >> 5;
  const int m = lane & 15;          // row within tile / N%16 for C frags
  const int g = lane >> 4;          // lane group (0/1)

  // Layer-1 / layer-2 weight fragments live in VGPRs for the whole kernel
  // (160 VGPRs; immune to the LDS-phase memory clobbers below).
  v16h w1[12], w2[8];
  #pragma unroll
  for (int f = 0; f < 12; ++f)
    w1[f] = *(const v16h*)(wpack + f * 512 + lane * 16);
  #pragma unroll
  for (int f = 0; f < 8; ++f)
    w2[f] = *(const v16h*)(wpack + L2_OFF + f * 512 + lane * 16);

  // Per-lane bias values, hoisted out of the tile loop.
  float bias1[4], bias2[4], bias3[2];
  #pragma unroll
  for (int nt = 0; nt < 4; ++nt) bias1[nt] = b1[nt * 16 + m];
  #pragma unroll
  for (int nt = 0; nt < 4; ++nt) bias2[nt] = b2[nt * 16 + m];
  #pragma unroll
  for (int nt = 0; nt < 2; ++nt) bias3[nt] = b3[nt * 16 + m];

  _Float16* xt = (_Float16*)slabs[wave];
  float*    of = (float*)slabs[wave];

  for (int tile = blockIdx.x * WAVES + wave; tile < n_tiles;
       tile += gridDim.x * WAVES) {
    const int base = tile << 4;

    // ---- gather: two atom rows per bond (lane -> row m, endpoint g) ----
    {
      int bi = base + m;
      int idx = g ? ba2[bi] : ba1[bi];
      const v4f* arow = (const v4f*)(atoms + (size_t)idx * 32);
      _Float16* drow = xt + m * XP + g * 32;
      #pragma unroll
      for (int i = 0; i < 8; ++i)
        *(v4h*)(drow + i * 4) = cvt4(arow[i]);
    }
    // ---- bond features: fully coalesced float4 ----
    {
      const v4f* brow = (const v4f*)(bonds + (size_t)base * 32);
      #pragma unroll
      for (int i = 0; i < 4; ++i) {
        int e = i * 32 + lane;            // 128 float4 per tile
        v4f v = brow[e];
        int r = e >> 3, c = (e & 7) << 2;
        *(v4h*)(xt + r * XP + 64 + c) = cvt4(v);
      }
    }
    asm volatile("s_wait_dscnt 0" ::: "memory");

    // ---- layer 1: [16x96]f16 x [96x64]f16 -> f32, B frags from registers ----
    V16U a1[3];
    #pragma unroll
    for (int kb = 0; kb < 3; ++kb) {
      a1[kb].h[0] = *(const v8h*)(xt + m * XP + kb * 32 + g * 8);
      a1[kb].h[1] = *(const v8h*)(xt + m * XP + kb * 32 + g * 8 + 16);
    }
    v8f c1[4];
    #pragma unroll
    for (int nt = 0; nt < 4; ++nt) {
      v8f acc = {};
      #pragma unroll
      for (int kb = 0; kb < 3; ++kb)
        acc = __builtin_amdgcn_wmma_f32_16x16x32_f16(
            false, a1[kb].v, false, w1[kb * 4 + nt], (short)0, acc, false, false);
      c1[nt] = acc;
    }
    #pragma unroll
    for (int nt = 0; nt < 4; ++nt) {
      #pragma unroll
      for (int r = 0; r < 8; ++r) {
        float v = rrelu(c1[nt][r] + bias1[nt]);
        xt[(r + 8 * g) * XP + nt * 16 + m] = (_Float16)v; // M = r+8g, N = nt*16+m
      }
    }
    asm volatile("s_wait_dscnt 0" ::: "memory");

    // ---- layer 2: [16x64] x [64x64], B frags from registers ----
    V16U a2[2];
    #pragma unroll
    for (int kb = 0; kb < 2; ++kb) {
      a2[kb].h[0] = *(const v8h*)(xt + m * XP + kb * 32 + g * 8);
      a2[kb].h[1] = *(const v8h*)(xt + m * XP + kb * 32 + g * 8 + 16);
    }
    v8f c2[4];
    #pragma unroll
    for (int nt = 0; nt < 4; ++nt) {
      v8f acc = {};
      #pragma unroll
      for (int kb = 0; kb < 2; ++kb)
        acc = __builtin_amdgcn_wmma_f32_16x16x32_f16(
            false, a2[kb].v, false, w2[kb * 4 + nt], (short)0, acc, false, false);
      c2[nt] = acc;
    }
    #pragma unroll
    for (int nt = 0; nt < 4; ++nt) {
      #pragma unroll
      for (int r = 0; r < 8; ++r) {
        float v = rrelu(c2[nt][r] + bias2[nt]);
        xt[(r + 8 * g) * XP + nt * 16 + m] = (_Float16)v;
      }
    }
    asm volatile("s_wait_dscnt 0" ::: "memory");

    // ---- layer 3: [16x64] x [64x32], B frags from LDS ----
    V16U a3[2];
    #pragma unroll
    for (int kb = 0; kb < 2; ++kb) {
      a3[kb].h[0] = *(const v8h*)(xt + m * XP + kb * 32 + g * 8);
      a3[kb].h[1] = *(const v8h*)(xt + m * XP + kb * 32 + g * 8 + 16);
    }
    #pragma unroll
    for (int nt = 0; nt < 2; ++nt) {
      v8f acc = {};
      #pragma unroll
      for (int kb = 0; kb < 2; ++kb) {
        v16h bf = *(const v16h*)(wlds3 + (kb * 2 + nt) * 512 + lane * 16);
        acc = __builtin_amdgcn_wmma_f32_16x16x32_f16(
            false, a3[kb].v, false, bf, (short)0, acc, false, false);
      }
      #pragma unroll
      for (int r = 0; r < 8; ++r)
        of[(r + 8 * g) * 32 + nt * 16 + m] = acc[r] + bias3[nt]; // f32 [16][32]
    }
    asm volatile("s_wait_dscnt 0" ::: "memory");

    // ---- coalesced output: 2048 B per tile as float4 ----
    {
      v4f* op = (v4f*)(out + (size_t)base * 32);
      const v4f* sf = (const v4f*)of;
      #pragma unroll
      for (int i = 0; i < 4; ++i) {
        int e = i * 32 + lane;
        op[e] = sf[e];
      }
    }
    asm volatile("s_wait_dscnt 0" ::: "memory");
  }
}

extern "C" void kernel_launch(void* const* d_in, const int* in_sizes, int n_in,
                              void* d_out, int out_size, void* d_ws, size_t ws_size,
                              hipStream_t stream) {
  const float* bonds = (const float*)d_in[0];
  const int*   ba1   = (const int*)d_in[1];
  const int*   ba2   = (const int*)d_in[2];
  const float* atoms = (const float*)d_in[3];
  const float* W1    = (const float*)d_in[4];
  const float* b1    = (const float*)d_in[5];
  const float* W2    = (const float*)d_in[6];
  const float* b2    = (const float*)d_in[7];
  const float* W3    = (const float*)d_in[8];
  const float* b3    = (const float*)d_in[9];
  float* out = (float*)d_out;
  _Float16* wpack = (_Float16*)d_ws;

  int n_bonds = in_sizes[0] / 32;
  int n_tiles = n_bonds / 16;    // N_BONDS = 1,600,000 -> exactly 100,000 tiles

  pack_weights_kernel<<<(W_TOTAL + 255) / 256, 256, 0, stream>>>(W1, W2, W3, wpack);

  int nblocks = (n_tiles + WAVES - 1) / WAVES;
  edge_mlp_kernel<<<nblocks, 256, 0, stream>>>(bonds, ba1, ba2, atoms, wpack,
                                               b1, b2, b3, out, n_tiles);
}
